// CoordinateDecoder_83038897701018
// MI455X (gfx1250) — matrix operand order repaired and checked by reading, not compiled, not run on hardware
//
#include <hip/hip_runtime.h>
#include <hip/hip_fp16.h>

typedef __attribute__((ext_vector_type(16))) _Float16 v16h;
typedef __attribute__((ext_vector_type(8)))  _Float16 v8h;
typedef __attribute__((ext_vector_type(8)))  float    v8f;

#define NPTS     65536
#define NBATCH   4
#define DM       128
#define GH       64
#define GW       64
#define MLPW     256
#define KPAD     256
#define ENCD     42
#define TILE_PTS 64
#define MTILES   4
#define HROW     264          // padded LDS row stride in halfs (bank-conflict-free A loads)
#define PI_F     3.14159265358979f

// overflow-safe fast tanh via hardware v_exp_f32
__device__ __forceinline__ float fast_tanh(float x) {
    float e = __expf(-2.0f * fabsf(x));
    float t = (1.0f - e) / (1.0f + e);
    return copysignf(t, x);
}

__device__ __forceinline__ float gelu_tanh(float x) {
    float x3 = x * x * x;
    float t  = fast_tanh(0.7978845608028654f * (x + 0.044715f * x3));
    return 0.5f * x * (1.0f + t);
}

// FiLM params: film[b][0:256] = gamma+1, film[b][256:512] = beta
__global__ void film_kernel(const float* __restrict__ ctx,
                            const float* __restrict__ w_film,
                            const float* __restrict__ b_film,
                            float* __restrict__ film) {
    int b = blockIdx.x;
    for (int j = threadIdx.x; j < 2 * MLPW; j += blockDim.x) {
        float s = b_film[j];
        for (int k = 0; k < DM; ++k)
            s += ctx[b * DM + k] * w_film[k * 2 * MLPW + j];
        film[b * 2 * MLPW + j] = s + (j < MLPW ? 1.0f : 0.0f);
    }
}

// Transpose + convert MLP weights to f16: wt[l][n][k], Kpad=256 (layer0 k>=170 zeroed)
__global__ void prep_weights(const float* __restrict__ w0,
                             const float* __restrict__ w1,
                             const float* __restrict__ w2,
                             const float* __restrict__ w3,
                             _Float16* __restrict__ wt) {
    int idx = blockIdx.x * blockDim.x + threadIdx.x;   // [0, 4*65536)
    int l = idx >> 16;
    int r = idx & 65535;
    int n = r >> 8;
    int k = r & 255;
    float v;
    if (l == 0)      v = (k < ENCD + DM) ? w0[k * MLPW + n] : 0.0f;
    else if (l == 1) v = w1[k * MLPW + n];
    else if (l == 2) v = w2[k * MLPW + n];
    else             v = w3[k * MLPW + n];
    wt[idx] = (_Float16)v;
}

// Output head weights: wt_out[n][k] for n in [0,16), cols >=3 zero-padded
__global__ void prep_wout(const float* __restrict__ w_out,
                          _Float16* __restrict__ wt_out) {
    int idx = blockIdx.x * blockDim.x + threadIdx.x;   // [0, 16*256)
    int n = idx >> 8;
    int k = idx & 255;
    wt_out[n * KPAD + k] = (_Float16)((n < 3) ? w_out[k * 3 + n] : 0.0f);
}

__global__ void __launch_bounds__(256)
decoder_kernel(const float* __restrict__ fg,
               const float* __restrict__ coords,
               const float* __restrict__ bm0,
               const float* __restrict__ bm1,
               const float* __restrict__ bm2,
               const float* __restrict__ bm3,
               const _Float16* __restrict__ wt,
               const _Float16* __restrict__ wt_out,
               const float* __restrict__ film,
               const float* __restrict__ b_out,
               float* __restrict__ out) {
    __shared__ _Float16 bufA[TILE_PTS * HROW];
    __shared__ _Float16 bufB[TILE_PTS * HROW];
    __shared__ float spy[TILE_PTS], spx[TILE_PTS];

    const int tid = threadIdx.x;
    const int b   = blockIdx.x >> 10;                 // 1024 blocks per batch image
    const int p0  = (blockIdx.x & 1023) * TILE_PTS;

    // ---- phase 0a: positional encoding + pixel coords ----
    if (tid < TILE_PTS) {
        int p = tid;
        float cx = coords[(p0 + p) * 2 + 0];
        float cy = coords[(p0 + p) * 2 + 1];
        _Float16* row = bufA + p * HROW;
        row[0] = (_Float16)cx;
        row[1] = (_Float16)cy;
        float fr = PI_F;
        #pragma unroll
        for (int f = 0; f < 10; ++f) {
            row[2 + 4 * f + 0] = (_Float16)__sinf(cx * fr);
            row[2 + 4 * f + 1] = (_Float16)__sinf(cy * fr);
            row[2 + 4 * f + 2] = (_Float16)__cosf(cx * fr);
            row[2 + 4 * f + 3] = (_Float16)__cosf(cy * fr);
            fr *= 2.0f;
        }
        for (int k = ENCD + DM; k < KPAD; ++k) row[k] = (_Float16)0.0f;
        spy[p] = (cx + 1.0f) * 0.5f * (float)(GH - 1);
        spx[p] = (cy + 1.0f) * 0.5f * (float)(GW - 1);
    }
    __syncthreads();

    // ---- phase 0b: bilinear sampling of feature grid ----
    for (int idx = tid; idx < TILE_PTS * DM; idx += 256) {
        int p = idx >> 7;
        int c = idx & (DM - 1);
        float py = spy[p], px = spx[p];
        float fy0 = floorf(py), fx0 = floorf(px);
        float wy = py - fy0, wx = px - fx0;
        int y0 = (int)fy0, x0 = (int)fx0;
        int y1 = y0 + 1, x1 = x0 + 1;
        y0 = min(max(y0, 0), GH - 1); y1 = min(max(y1, 0), GH - 1);
        x0 = min(max(x0, 0), GW - 1); x1 = min(max(x1, 0), GW - 1);
        const float* base = fg + (size_t)b * GH * GW * DM;
        float v00 = base[((y0 * GW) + x0) * DM + c];
        float v01 = base[((y0 * GW) + x1) * DM + c];
        float v10 = base[((y1 * GW) + x0) * DM + c];
        float v11 = base[((y1 * GW) + x1) * DM + c];
        float v = (1.0f - wy) * ((1.0f - wx) * v00 + wx * v01)
                +         wy  * ((1.0f - wx) * v10 + wx * v11);
        bufA[p * HROW + ENCD + c] = (_Float16)v;
    }
    __syncthreads();

    // ---- MLP layers via WMMA ----
    const int lane    = tid & 31;
    const int wv      = tid >> 5;
    const int col     = lane & 15;
    const int halfsel = lane >> 4;       // 0: lanes 0-15, 1: lanes 16-31
    const int nbase   = wv * 32;         // each wave owns 32 output columns

    const float* biases[4] = { bm0, bm1, bm2, bm3 };
    const _Float16* hcur = bufA;
    _Float16*       hnxt = bufB;

    #pragma unroll
    for (int l = 0; l < 4; ++l) {
        const _Float16* W = wt + l * (MLPW * KPAD);
        float bias0 = biases[l][nbase + col];
        float bias1 = biases[l][nbase + 16 + col];
        v8f acc[MTILES][2];
        #pragma unroll
        for (int mt = 0; mt < MTILES; ++mt)
            #pragma unroll
            for (int r = 0; r < 8; ++r) { acc[mt][0][r] = bias0; acc[mt][1][r] = bias1; }

        #pragma unroll
        for (int ks = 0; ks < 8; ++ks) {
            // B fragment: element e -> K = ks*32 + halfsel*16 + e, N = nbase(+16) + col
            const int kbB = ks * 32 + halfsel * 16;
            v16h bf0, bf1;
            {
                const v8h* pb0 = (const v8h*)(W + (nbase + col) * KPAD + kbB);
                const v8h* pb1 = (const v8h*)(W + (nbase + 16 + col) * KPAD + kbB);
                ((v8h*)&bf0)[0] = pb0[0]; ((v8h*)&bf0)[1] = pb0[1];
                ((v8h*)&bf1)[0] = pb1[0]; ((v8h*)&bf1)[1] = pb1[1];
            }
            // A fragment: element e -> K = kbA + e (kbA = ks*32 + halfsel*16)
            const int kbA = ks * 32 + halfsel * 16;
            #pragma unroll
            for (int mt = 0; mt < MTILES; ++mt) {
                const _Float16* ap = hcur + (mt * 16 + col) * HROW + kbA;
                v16h af;
                ((v8h*)&af)[0] = *(const v8h*)ap;
                ((v8h*)&af)[1] = *(const v8h*)(ap + 8);
                acc[mt][0] = __builtin_amdgcn_wmma_f32_16x16x32_f16(
                    false, af, false, bf0, (short)0, acc[mt][0], false, false);
                acc[mt][1] = __builtin_amdgcn_wmma_f32_16x16x32_f16(
                    false, af, false, bf1, (short)0, acc[mt][1], false, false);
            }
        }

        // FiLM + GELU, write next activation tile
        const float g0  = film[b * 512 + nbase + col];
        const float g1  = film[b * 512 + nbase + 16 + col];
        const float be0 = film[b * 512 + 256 + nbase + col];
        const float be1 = film[b * 512 + 256 + nbase + 16 + col];
        #pragma unroll
        for (int mt = 0; mt < MTILES; ++mt) {
            #pragma unroll
            for (int r = 0; r < 8; ++r) {
                int row = mt * 16 + r + halfsel * 8;   // D layout: M = r (+8 for hi lanes)
                float v0 = gelu_tanh(acc[mt][0][r] * g0 + be0);
                float v1 = gelu_tanh(acc[mt][1][r] * g1 + be1);
                hnxt[row * HROW + nbase + col]      = (_Float16)v0;
                hnxt[row * HROW + nbase + 16 + col] = (_Float16)v1;
            }
        }
        __syncthreads();
        const _Float16* t = hcur; hcur = hnxt; hnxt = (_Float16*)t;
    }

    // ---- output head via WMMA: 256 -> 16 (cols 0..2 valid), tanh ----
    if (wv < MTILES) {                       // wave-uniform: EXEC all-1s inside
        const int mt = wv;
        v8f acc;
        float hb = (col < 3) ? b_out[col] : 0.0f;
        #pragma unroll
        for (int r = 0; r < 8; ++r) acc[r] = hb;
        #pragma unroll
        for (int ks = 0; ks < 8; ++ks) {
            const int kbB = ks * 32 + halfsel * 16;
            v16h bf;
            const v8h* pb = (const v8h*)(wt_out + col * KPAD + kbB);
            ((v8h*)&bf)[0] = pb[0]; ((v8h*)&bf)[1] = pb[1];
            const int kbA = ks * 32 + halfsel * 16;
            const _Float16* ap = hcur + (mt * 16 + col) * HROW + kbA;
            v16h af;
            ((v8h*)&af)[0] = *(const v8h*)ap;
            ((v8h*)&af)[1] = *(const v8h*)(ap + 8);
            acc = __builtin_amdgcn_wmma_f32_16x16x32_f16(
                false, af, false, bf, (short)0, acc, false, false);
        }
        if (col < 3) {
            #pragma unroll
            for (int r = 0; r < 8; ++r) {
                int p = mt * 16 + r + halfsel * 8;
                out[((size_t)b * NPTS + p0 + p) * 3 + col] = fast_tanh(acc[r]);
            }
        }
    }
}

extern "C" void kernel_launch(void* const* d_in, const int* in_sizes, int n_in,
                              void* d_out, int out_size, void* d_ws, size_t ws_size,
                              hipStream_t stream) {
    (void)in_sizes; (void)n_in; (void)out_size; (void)ws_size;
    const float* fg     = (const float*)d_in[0];
    const float* ctx    = (const float*)d_in[1];
    const float* coords = (const float*)d_in[2];
    const float* w0 = (const float*)d_in[3];  const float* b0 = (const float*)d_in[4];
    const float* w1 = (const float*)d_in[5];  const float* b1 = (const float*)d_in[6];
    const float* w2 = (const float*)d_in[7];  const float* b2 = (const float*)d_in[8];
    const float* w3 = (const float*)d_in[9];  const float* b3 = (const float*)d_in[10];
    const float* wf = (const float*)d_in[11]; const float* bf = (const float*)d_in[12];
    const float* wo = (const float*)d_in[13]; const float* bo = (const float*)d_in[14];

    float*    film  = (float*)d_ws;                              // 4*512 floats  (8 KB)
    _Float16* wt    = (_Float16*)((char*)d_ws + 8192);           // 4*256*256 f16 (512 KB)
    _Float16* wtout = (_Float16*)((char*)d_ws + 8192 + 524288);  // 16*256 f16    (8 KB)

    film_kernel<<<NBATCH, 256, 0, stream>>>(ctx, wf, bf, film);
    prep_weights<<<(4 * MLPW * KPAD) / 256, 256, 0, stream>>>(w0, w1, w2, w3, wt);
    prep_wout<<<16, 256, 0, stream>>>(wo, wtout);
    decoder_kernel<<<NBATCH * (NPTS / TILE_PTS), 256, 0, stream>>>(
        fg, coords, b0, b1, b2, b3, wt, wtout, film, bo, (float*)d_out);
}
